// MotionGenerator_11656541241920
// MI455X (gfx1250) — compile-verified, matrix operand order checked by reading, not run on hardware
//
#include <hip/hip_runtime.h>
#include <hip/hip_bf16.h>

// ---------------------------------------------------------------------------
// Problem constants (reference setup: N=512 points, W=L=2048, sigma=32, mag=1)
// ---------------------------------------------------------------------------
#define NPTS 512
#define WDIM 2048
#define LDIM 2048

typedef _Float16 h16 __attribute__((ext_vector_type(16)));
typedef _Float16 h8  __attribute__((ext_vector_type(8)));
typedef float    f8  __attribute__((ext_vector_type(8)));

// ---------------------------------------------------------------------------
// Phase 1: build f16 operand tables in workspace.
//   A[c][x][n] = exp(-(x - xc[n])^2 / (2*sigma^2)) * v[c][n] * scale
//   B[y][n]    = exp(-(y - yc[n])^2 / (2*sigma^2))            (this is B^T)
// One thread per (x, n); since W == L it also fills row y=x of B.
// ---------------------------------------------------------------------------
__global__ void __launch_bounds__(256)
build_tables_kernel(const float* __restrict__ v,   // [2][NPTS]
                    const int*   __restrict__ xc,  // [NPTS]
                    const int*   __restrict__ yc,  // [NPTS]
                    _Float16*    __restrict__ A,   // [2][WDIM][NPTS]
                    _Float16*    __restrict__ B)   // [LDIM][NPTS]
{
    int idx = blockIdx.x * blockDim.x + threadIdx.x;
    int n = idx & (NPTS - 1);
    int x = idx >> 9;                 // NPTS = 512 = 2^9
    if (x >= WDIM) return;

    const float inv2s2 = 1.0f / (2.0f * 32.0f * 32.0f);         // sigma = 32
    const float scale  = 32.0f / (float)(WDIM + LDIM);          // magnitude=1

    float fx = (float)x;
    float dx = fx - (float)xc[n];
    float wx = expf(-dx * dx * inv2s2);
    float dy = fx - (float)yc[n];
    float wy = expf(-dy * dy * inv2s2);

    float v0 = v[n]        * scale;
    float v1 = v[NPTS + n] * scale;

    A[(size_t)x * NPTS + n]                        = (_Float16)(wx * v0);
    A[(size_t)(WDIM + x) * NPTS + n]               = (_Float16)(wx * v1);
    B[(size_t)x * NPTS + n]                        = (_Float16)wy;
}

// ---------------------------------------------------------------------------
// Fragment loader for the CDNA5 16-bit 16x32 A-matrix layout (ISA 7.12.2):
//   lanes 0-15  hold row M=lane,    K = 0..7  in regs 0..3, K = 16..23 in 4..7
//   lanes 16-31 hold row M=lane-16, K = 8..15 in regs 0..3, K = 24..31 in 4..7
// Caller passes p = &row[k_base + khalf] where khalf = (lane>=16) ? 8 : 0.
// B^T rows loaded the same way (lane = output column).
// ---------------------------------------------------------------------------
__device__ __forceinline__ h16 load_frag16(const _Float16* __restrict__ p) {
    h8 lo = *(const h8*)(p);        // K offsets +0..+7   (16B aligned)
    h8 hi = *(const h8*)(p + 16);   // K offsets +16..+23
    h16 r;
#pragma unroll
    for (int i = 0; i < 8; ++i) { r[i] = lo[i]; r[8 + i] = hi[i]; }
    return r;
}

// ---------------------------------------------------------------------------
// Phase 2: field[c] = A_c (2048x512, f16) x B^T (512x2048, f16) -> f32.
// One wave computes a 32x64 output tile: 2 A-frags x 4 B-frags,
// 8 x v_wmma_f32_16x16x32_f16 per k-step, K = 512 -> 16 k-steps.
// ---------------------------------------------------------------------------
__global__ void __launch_bounds__(256)
wmma_field_kernel(const _Float16* __restrict__ A,  // [2][WDIM][NPTS]
                  const _Float16* __restrict__ B,  // [LDIM][NPTS] (= B^T)
                  float*          __restrict__ out)// [2][WDIM][LDIM]
{
    const int K = NPTS;
    int wave = (int)((blockIdx.x * blockDim.x + threadIdx.x) >> 5);
    int lane = (int)(threadIdx.x & 31);

    int nt = wave & 31;          // 32 n-supertiles of 64 columns
    int mt = (wave >> 5) & 63;   // 64 m-supertiles of 32 rows
    int c  = wave >> 11;         // channel 0/1

    int rlane = lane & 15;
    int khalf = (lane >> 4) << 3;       // 0 or 8 (A/B fragment K-half)

    const _Float16* Abase = A + (size_t)c * WDIM * K;
    int m0    = mt * 32 + rlane;
    int nbase = nt * 64;

    const _Float16* arow0 = Abase + (size_t)m0 * K + khalf;
    const _Float16* arow1 = arow0 + (size_t)16 * K;
    const _Float16* brow  = B + (size_t)(nbase + rlane) * K + khalf;

    f8 acc[2][4];
#pragma unroll
    for (int mi = 0; mi < 2; ++mi)
#pragma unroll
        for (int j = 0; j < 4; ++j)
            acc[mi][j] = (f8)0.0f;

#pragma unroll 2
    for (int kk = 0; kk < K; kk += 32) {
        h16 a0 = load_frag16(arow0 + kk);
        h16 a1 = load_frag16(arow1 + kk);
        h16 bf[4];
#pragma unroll
        for (int j = 0; j < 4; ++j)
            bf[j] = load_frag16(brow + kk + (size_t)j * 16 * K);

#pragma unroll
        for (int j = 0; j < 4; ++j) {
            acc[0][j] = __builtin_amdgcn_wmma_f32_16x16x32_f16(
                false, a0, false, bf[j], (short)0, acc[0][j], false, false);
            acc[1][j] = __builtin_amdgcn_wmma_f32_16x16x32_f16(
                false, a1, false, bf[j], (short)0, acc[1][j], false, false);
        }
    }

    // C/D layout: VGPR r -> row M = r (+8 for lanes 16-31), column = lane&15.
    int rowoff = (lane >> 4) << 3;      // 0 or 8
#pragma unroll
    for (int mi = 0; mi < 2; ++mi) {
#pragma unroll
        for (int j = 0; j < 4; ++j) {
            float* o = out + (size_t)c * WDIM * LDIM
                           + (size_t)(mt * 32 + mi * 16 + rowoff) * LDIM
                           + (size_t)(nbase + j * 16 + rlane);
#pragma unroll
            for (int r = 0; r < 8; ++r)
                o[(size_t)r * LDIM] = acc[mi][j][r];
        }
    }
}

// ---------------------------------------------------------------------------
// Launch: inputs are (init_vectors f32[2,512], x_coord i32[512],
// y_coord i32[512], width, lenth). Output f32 [1,2,2048,2048].
// Workspace: A tables (4 MB f16) + B table (2 MB f16).
// ---------------------------------------------------------------------------
extern "C" void kernel_launch(void* const* d_in, const int* in_sizes, int n_in,
                              void* d_out, int out_size, void* d_ws, size_t ws_size,
                              hipStream_t stream) {
    (void)in_sizes; (void)n_in; (void)out_size; (void)ws_size;

    const float* v  = (const float*)d_in[0];
    const int*   xc = (const int*)d_in[1];
    const int*   yc = (const int*)d_in[2];

    _Float16* A = (_Float16*)d_ws;                       // [2][WDIM][NPTS]
    _Float16* B = A + (size_t)2 * WDIM * NPTS;           // [LDIM][NPTS]
    float*  out = (float*)d_out;

    // Phase 1: WDIM*NPTS threads.
    int blocks1 = (WDIM * NPTS) / 256;                   // 4096
    build_tables_kernel<<<blocks1, 256, 0, stream>>>(v, xc, yc, A, B);

    // Phase 2: 2 * 64 * 32 = 4096 waves, 8 waves per 256-thread block.
    int totalWaves = 2 * 64 * 32;
    int blocks2 = totalWaves / 8;                        // 512
    wmma_field_kernel<<<blocks2, 256, 0, stream>>>(A, B, out);
}
